// LocalPolarHead_36318243455492
// MI455X (gfx1250) — matrix operand order repaired and checked by reading, not compiled, not run on hardware
//
#include <hip/hip_runtime.h>
#include <math.h>

typedef __attribute__((ext_vector_type(16))) _Float16 v16h;
typedef __attribute__((ext_vector_type(8)))  float    v8f;
typedef __attribute__((ext_vector_type(4)))  unsigned int v4u;
typedef __attribute__((ext_vector_type(8)))  int      v8i_;
typedef __attribute__((ext_vector_type(4)))  int      v4i_;

#define C64   64
#define FH    12
#define FW    30
#define PH_   96
#define PW_   240
#define NPRI  (PH_ * PW_)      // 23040
#define NB    16
#define TOPK  4096
#define PI_F  3.14159265358979f

union Frag16 { v16h v; _Float16 h[16]; };

// ---------------------------------------------------------------------------
// Fused head. Per workgroup: TDM-stage a 3-src-row x 64-channel f32 slab of
// feats into LDS, then bilinear -> WMMA 64x64 conv -> BN/ReLU -> cls dot ->
// atan reg -> polar_map. One wave handles 16 pixels x 64 channels per step.
// ---------------------------------------------------------------------------
__global__ __launch_bounds__(256)
void polar_head_kernel(const float* __restrict__ feats,
                       const float* __restrict__ reg_w,   // [2,64]
                       const float* __restrict__ w1,      // [64,64]
                       const float* __restrict__ b1,      // [64]
                       const float* __restrict__ bn_g,
                       const float* __restrict__ bn_b,
                       const float* __restrict__ bn_m,
                       const float* __restrict__ bn_v,
                       const float* __restrict__ w2,      // [1,64]
                       const float* __restrict__ b2p,     // [1]
                       float* __restrict__ out)
{
    __shared__ float sfeat[C64 * 3 * FW];   // 64 ch x 3 src rows x 30 = 23040 B
    __shared__ float srw[2 * C64];          // reg weights

    const int tid  = threadIdx.x;
    const int wg   = blockIdx.x;
    const int b    = wg / 12;
    const int blk  = wg % 12;               // 12 blocks of 8 dest rows (1920 px)
    const int FSZ  = C64 * FH * FW;         // 23040

    // source rows needed by dest rows [blk*8, blk*8+8): fy = y*11/95 spans
    // < 1 src row, so rows [ys, ys+2] always suffice (ys clamped to FH-3).
    const int ys = min((blk * 88) / 95, FH - 3);

    // Workgroup-relative LDS byte offset of sfeat: generic LDS pointers carry
    // the offset in addr[31:0] (flat aperture rules).
    const unsigned lds_off = (unsigned)(uintptr_t)(void*)&sfeat[0];

    // ---- TDM: 64 lines x 90 contiguous f32, line stride 360 f32 ----------
    if (tid == 0) {
        const unsigned long long ga =
            (unsigned long long)(uintptr_t)(feats + (size_t)b * FSZ + ys * FW);
        const v4u g0 = {
            1u,                                  // count=1 (valid user D#)
            lds_off,                             // lds_addr (bytes)
            (unsigned)(ga & 0xFFFFFFFFu),        // global_addr[31:0]
            (unsigned)(((ga >> 32) & 0x01FFFFFFu) | 0x80000000u) // [56:32]|type=2
        };
        const v8i_ g1 = {
            (int)0x00020000u,                    // mask=0, data_size=2 (4B)
            (int)(90u << 16),                    // tensor_dim0 = 90 (lo16 here)
            (int)(64u << 16),                    // dim0 hi=0 | tensor_dim1 = 64
            (int)(90u << 16),                    // dim1 hi=0 | tile_dim0 = 90
            (int)64,                             // tile_dim1 = 64 | tile_dim2=0
            (int)360,                            // tensor_dim0_stride = 360
            0, 0                                 // stride0 hi | tensor_dim1_stride
        };
        const v4i_ gz = { 0, 0, 0, 0 };
#if defined(__clang_major__) && __clang_major__ >= 23
        const v8i_ gz8 = { 0, 0, 0, 0, 0, 0, 0, 0 };
        __builtin_amdgcn_tensor_load_to_lds(g0, g1, gz, gz, gz8, 0);
#else
        __builtin_amdgcn_tensor_load_to_lds(g0, g1, gz, gz, 0);
#endif
    }
    for (int i = tid; i < 2 * C64; i += 256) srw[i] = reg_w[i];
    __builtin_amdgcn_s_wait_tensorcnt(0);
    // The TDM intrinsic's LDS writes are invisible to alias analysis: force
    // the compiler to treat sfeat as externally modified so the interp loads
    // below are not folded to undef.
    asm volatile("" : : "r"(&sfeat[0]) : "memory");
    __syncthreads();

    const float* rw0 = srw;
    const float* rw1 = srw + C64;

    const int lane = tid & 31;
    const int wave = tid >> 5;
    const int lm   = lane & 15;     // pixel-in-group (A rows / C columns)
    const int lh   = lane >> 4;     // lane half -> K sub-chunk

    // ---- B fragments (weights as f16, resident for whole kernel) ----------
    // B layout used here (contiguous k, same as A): lane = column N (= output channel), lanes 0-15 hold
    // K=0..15, lanes 16-31 hold K=16..31 of each 32-deep k-tile.
    Frag16 bf[2][4];
#pragma unroll
    for (int kk = 0; kk < 2; ++kk) {
#pragma unroll
        for (int nn = 0; nn < 4; ++nn) {
            const int o  = nn * 16 + lm;
            const int cb = kk * 32 + lh * 16;
            const float* row = w1 + o * C64 + cb;
#pragma unroll
            for (int j = 0; j < 16; ++j) bf[kk][nn].h[j] = (_Float16)row[j];
        }
    }

    // ---- fold conv bias + BN into h*s + t, and grab w2 per lane ----------
    float sC[4], tC[4], w2l[4];
#pragma unroll
    for (int nn = 0; nn < 4; ++nn) {
        const int o  = nn * 16 + lm;
        const float inv = rsqrtf(bn_v[o] + 1e-5f);
        const float g   = bn_g[o];
        sC[nn]  = g * inv;
        tC[nn]  = (b1[o] - bn_m[o]) * g * inv + bn_b[o];
        w2l[nn] = w2[o];
    }
    const float bias2 = b2p[0];

    const int  blockBase = blk * 1920;
    float* outB = out + (size_t)b * 3 * NPRI;

    for (int t = 0; t < 15; ++t) {
        const int gstart = blockBase + t * 128 + wave * 16; // 16-aligned: one row
        const int p = gstart + lm;
        const int y = p / PW_;
        const int x = p - y * PW_;

        // align_corners bilinear source coordinates
        const float fy = (float)y * ((float)(FH - 1) / (float)(PH_ - 1));
        const float fx = (float)x * ((float)(FW - 1) / (float)(PW_ - 1));
        const int y0 = (int)fy; const int y1 = min(y0 + 1, FH - 1);
        const int x0 = (int)fx; const int x1 = min(x0 + 1, FW - 1);
        const float wy = fy - (float)y0, wx = fx - (float)x0;
        const float wyc = 1.0f - wy, wxc = 1.0f - wx;
        const int ly0 = y0 - ys, ly1 = y1 - ys;   // 0..2 within the LDS slab

        v8f acc[4] = {};
        float r0 = 0.0f, r1 = 0.0f;

#pragma unroll
        for (int kk = 0; kk < 2; ++kk) {
            Frag16 a;
            // A layout: lane lm = row M; element j -> K = lh*16 + j
#pragma unroll
            for (int j = 0; j < 16; ++j) {
                const int c = kk * 32 + lh * 16 + j;
                const float* fp = sfeat + c * (3 * FW);
                const float v00 = fp[ly0 * FW + x0];
                const float v01 = fp[ly0 * FW + x1];
                const float v10 = fp[ly1 * FW + x0];
                const float v11 = fp[ly1 * FW + x1];
                const float xv = (v00 * wxc + v01 * wx) * wyc
                               + (v10 * wxc + v11 * wx) * wy;
                a.h[j] = (_Float16)xv;
                r0 = fmaf(xv, rw0[c], r0);
                r1 = fmaf(xv, rw1[c], r1);
            }
#pragma unroll
            for (int nn = 0; nn < 4; ++nn) {
                acc[nn] = __builtin_amdgcn_wmma_f32_16x16x32_f16(
                    false, a.v, false, bf[kk][nn].v,
                    (short)0, acc[nn], false, false);
            }
        }

        // BN + ReLU + cls dot. C layout: lane holds channel o=nn*16+lm for
        // pixels M = r + 8*lh (r = accumulator index).
        float clsp[8];
#pragma unroll
        for (int r = 0; r < 8; ++r) clsp[r] = 0.0f;
#pragma unroll
        for (int nn = 0; nn < 4; ++nn) {
#pragma unroll
            for (int r = 0; r < 8; ++r) {
                float hv = fmaf(acc[nn][r], sC[nn], tC[nn]);
                hv = fmaxf(hv, 0.0f);
                clsp[r] = fmaf(hv, w2l[nn], clsp[r]);
            }
        }
        // reduce over the 16 N-lanes within each half
#pragma unroll
        for (int r = 0; r < 8; ++r) {
            clsp[r] += __shfl_xor(clsp[r], 1, 32);
            clsp[r] += __shfl_xor(clsp[r], 2, 32);
            clsp[r] += __shfl_xor(clsp[r], 4, 32);
            clsp[r] += __shfl_xor(clsp[r], 8, 32);
        }
        // route pixel (lane&15)'s cls back to its lane
        const int rs = lane & 7;
        float v = clsp[0];
#pragma unroll
        for (int r = 1; r < 8; ++r) v = (rs == r) ? clsp[r] : v;
        const float v2 = __shfl_xor(v, 24, 32);
        float clsv = ((((lane >> 3) & 1) == lh) ? v : v2) + bias2;

        // reg dots: lane halves partition the 64 channels
        const float fr0 = r0 + __shfl_xor(r0, 16, 32);
        const float fr1 = r1 + __shfl_xor(r1, 16, 32);
        const float ra  = atanf(fr0) * (1.0f / PI_F);
        const float rr  = atanf(fr1) * (1.0f / PI_F);

        if (lane < 16) {
            outB[p]             = clsv;
            outB[NPRI + p]      = ra;
            outB[2 * NPRI + p]  = rr;
        }
    }
}

// ---------------------------------------------------------------------------
// Deterministic per-batch top-4096: radix-threshold (1024-bin histogram on
// sign-flipped float keys) + ordered ballot-scan compaction, recomputing the
// embedding geometry from polar_map at emit time.
// ---------------------------------------------------------------------------
__device__ inline unsigned fkey(float f) {
    unsigned u = __float_as_uint(f);
    return (u & 0x80000000u) ? ~u : (u | 0x80000000u);
}

__device__ inline void emit(const float* __restrict__ cplane,
                            float* __restrict__ embBase,
                            int* __restrict__ idBase,
                            unsigned slot, int p) {
    const float ra = cplane[NPRI + p];
    const float rr = cplane[2 * NPRI + p];
    const float angle = ra * PI_F;
    const float lrho  = rr * (800.0f / 240.0f);
    const int y = p / PW_, x = p - y * PW_;
    const float u  = ((float)x + 0.5f) * (800.0f / 240.0f);
    const float vv = ((float)y + 0.5f) * (320.0f / 96.0f);
    const float gx = u - 400.0f, gy = 320.0f - vv;
    float sn, cs;
    __sincosf(angle, &sn, &cs);
    const float grho = lrho + gx * cs + gy * sn;
    embBase[slot * 2]     = fminf(fmaxf(ra, -0.45f), 0.45f);
    embBase[slot * 2 + 1] = grho * (1.0f / 800.0f);
    idBase[slot] = p;
}

__global__ __launch_bounds__(256)
void topk_kernel(float* __restrict__ out)
{
    __shared__ unsigned hist[1024];
    __shared__ unsigned wtot[8];
    __shared__ unsigned shT, shAbove, shBase;

    const int b    = blockIdx.x;
    const int tid  = threadIdx.x;
    const int lane = tid & 31;
    const int wave = tid >> 5;

    const float* cplane = out + (size_t)b * 3 * NPRI;
    float* embBase = out + (size_t)NB * 3 * NPRI + (size_t)b * TOPK * 2;
    int*   idBase  = (int*)(out + (size_t)NB * 3 * NPRI + (size_t)NB * TOPK * 2)
                     + b * TOPK;

    for (int i = tid; i < TOPK * 2; i += 256) embBase[i] = 0.0f;
    for (int i = tid; i < TOPK; i += 256) idBase[i] = 0;
    for (int i = tid; i < 1024; i += 256) hist[i] = 0;
    __syncthreads();

    for (int i = tid; i < NPRI; i += 256)
        atomicAdd(&hist[fkey(cplane[i]) >> 22], 1u);
    __syncthreads();

    if (tid == 0) {
        unsigned cum = 0, Tb = 0;
        for (int bin = 1023; bin >= 0; --bin) {
            const unsigned h = hist[bin];
            if (cum + h >= TOPK) { Tb = (unsigned)bin; break; }
            cum += h;
        }
        shT = Tb; shAbove = cum; shBase = 0;
    }
    __syncthreads();
    const unsigned Tb = shT, above = shAbove;

    // pass 1: strictly-above-threshold bins, index-ordered compaction
    for (int start = 0; start < NPRI; start += 256) {
        const int i = start + tid;
        const bool pred = (i < NPRI) && ((fkey(cplane[i]) >> 22) > Tb);
        const unsigned m   = (unsigned)__ballot(pred);
        const unsigned pre = __popc(m & ((1u << lane) - 1u));
        if (lane == 0) wtot[wave] = __popc(m);
        __syncthreads();
        unsigned off = pre, ctot = 0;
        for (int w = 0; w < 8; ++w) { if (w < wave) off += wtot[w]; ctot += wtot[w]; }
        const unsigned slot = shBase + off;
        if (pred) emit(cplane, embBase, idBase, slot, i);
        __syncthreads();
        if (tid == 0) shBase += ctot;
        __syncthreads();
    }

    if (tid == 0) shBase = 0;
    __syncthreads();

    // pass 2: equal-to-threshold bin, capped at TOPK
    for (int start = 0; start < NPRI; start += 256) {
        const int i = start + tid;
        const bool pred = (i < NPRI) && ((fkey(cplane[i]) >> 22) == Tb);
        const unsigned m   = (unsigned)__ballot(pred);
        const unsigned pre = __popc(m & ((1u << lane) - 1u));
        if (lane == 0) wtot[wave] = __popc(m);
        __syncthreads();
        unsigned off = pre, ctot = 0;
        for (int w = 0; w < 8; ++w) { if (w < wave) off += wtot[w]; ctot += wtot[w]; }
        const unsigned slot = above + shBase + off;
        if (pred && slot < TOPK) emit(cplane, embBase, idBase, slot, i);
        __syncthreads();
        if (tid == 0) shBase += ctot;
        __syncthreads();
    }
}

// ---------------------------------------------------------------------------
extern "C" void kernel_launch(void* const* d_in, const int* in_sizes, int n_in,
                              void* d_out, int out_size, void* d_ws, size_t ws_size,
                              hipStream_t stream) {
    (void)in_sizes; (void)n_in; (void)out_size; (void)d_ws; (void)ws_size;
    const float* feats = (const float*)d_in[0];
    const float* reg_w = (const float*)d_in[1];
    const float* w1    = (const float*)d_in[2];
    const float* b1    = (const float*)d_in[3];
    const float* bn_g  = (const float*)d_in[4];
    const float* bn_b  = (const float*)d_in[5];
    const float* bn_m  = (const float*)d_in[6];
    const float* bn_v  = (const float*)d_in[7];
    const float* w2    = (const float*)d_in[8];
    const float* b2    = (const float*)d_in[9];
    float* out = (float*)d_out;

    polar_head_kernel<<<dim3(NB * 12), dim3(256), 0, stream>>>(
        feats, reg_w, w1, b1, bn_g, bn_b, bn_m, bn_v, w2, b2, out);
    topk_kernel<<<dim3(NB), dim3(256), 0, stream>>>(out);
}
